// DCRNN_76725295776119
// MI455X (gfx1250) — matrix operand order repaired and checked by reading, not compiled, not run on hardware
//
#include <hip/hip_runtime.h>
#include <hip/hip_bf16.h>
#include <math.h>

#define NNODES 50000
#define NEDGES 500000
#define FDIM   64
#define CDIM   128
#define NSRC   5
#define KTOT   (NSRC*CDIM)     // 640
#define NTILES (NNODES/16)     // 3125 (exact)

typedef __attribute__((ext_vector_type(16))) __bf16 v16bf;
typedef __attribute__((ext_vector_type(8)))  float  v8f;

static inline int cdiv_host(long long a, long long b){ return (int)((a+b-1)/b); }

// ---------------- utility kernels ----------------

__global__ void k_zero2(float* __restrict__ a, float* __restrict__ b, int n) {
  int i = blockIdx.x*blockDim.x + threadIdx.x;
  if (i < n) { a[i] = 0.f; b[i] = 0.f; }
}

__global__ void k_negcopy2(float* __restrict__ a, float* __restrict__ b,
                           const float* __restrict__ src, int n) {
  int i = blockIdx.x*blockDim.x + threadIdx.x;
  if (i < n) { float v = -src[i]; a[i] = v; b[i] = v; }
}

__global__ void k_degree(const int* __restrict__ row, const int* __restrict__ col,
                         float* __restrict__ deg_o, float* __restrict__ deg_i, int e) {
  int i = blockIdx.x*blockDim.x + threadIdx.x;
  if (i < e) {
    atomicAdd(&deg_o[row[i]], 1.f);
    atomicAdd(&deg_i[col[i]], 1.f);
  }
}

__global__ void k_norm(const int* __restrict__ row,
                       const float* __restrict__ deg_o, const float* __restrict__ deg_i,
                       float* __restrict__ no, float* __restrict__ ni, int e) {
  int i = blockIdx.x*blockDim.x + threadIdx.x;
  if (i < e) { int r = row[i]; no[i] = 1.f/deg_o[r]; ni[i] = 1.f/deg_i[r]; }
}

__global__ void k_build_xh(const float* __restrict__ X, const float* __restrict__ H,
                           float* __restrict__ XH, int n /* NNODES*FDIM */) {
  int i = blockIdx.x*blockDim.x + threadIdx.x;
  if (i < n) {
    int nd = i >> 6, f = i & 63;
    XH[nd*CDIM + f]        = X[i];
    XH[nd*CDIM + FDIM + f] = H[i];
  }
}

__global__ void k_build_xhr(const float* __restrict__ H, const float* __restrict__ R,
                            float* __restrict__ XH, int n /* NNODES*FDIM */) {
  int i = blockIdx.x*blockDim.x + threadIdx.x;
  if (i < n) {
    int nd = i >> 6, f = i & 63;
    XH[nd*CDIM + FDIM + f] = H[i]*R[i];  // X half already in place
  }
}

// Build transposed bf16 weight panels Wt[c][s*128+k], c in [0,64), with
// s=0 -> W[0,0]+W[1,0]; s=1 -> W[0,1]; s=2 -> W[1,1]; s=3 -> W[0,2]; s=4 -> W[1,2]
__global__ void k_prep_weights(const float* __restrict__ Wz, const float* __restrict__ Wr,
                               const float* __restrict__ Wh,
                               __bf16* __restrict__ tz, __bf16* __restrict__ tr,
                               __bf16* __restrict__ th) {
  int t = blockIdx.x*blockDim.x + threadIdx.x;
  if (t >= FDIM*KTOT) return;
  int c  = t / KTOT;
  int sk = t - c*KTOT;
  int s  = sk >> 7;
  int k  = sk & 127;
  int d0, k0; float m1;
  if (s == 0) { d0 = 0; k0 = 0; m1 = 1.f; }
  else        { d0 = (s-1)&1; k0 = (s+1)>>1; m1 = 0.f; }
  int i0 = ((d0*3 + k0)*CDIM + k)*FDIM + c;          // selected block
  int i1 = ((1*3 + 0)*CDIM + k)*FDIM + c;            // W[1,0] (only used when s==0)
  tz[t] = (__bf16)(Wz[i0] + m1*Wz[i1]);
  tr[t] = (__bf16)(Wr[i0] + m1*Wr[i1]);
  th[t] = (__bf16)(Wh[i0] + m1*Wh[i1]);
}

// Fused first-order propagation: one gather of in[row], two scatters.
// outA[col[e]] += nA[e]*in[row[e]] ; outB[col[e]] += nB[e]*in[row[e]]
__global__ void k_scatter2(const float* __restrict__ in,
                           float* __restrict__ outA, float* __restrict__ outB,
                           const int* __restrict__ row, const int* __restrict__ col,
                           const float* __restrict__ nA, const float* __restrict__ nB,
                           int e) {
  int t = blockIdx.x*blockDim.x + threadIdx.x;
  int ed = __builtin_amdgcn_readfirstlane(t >> 5);   // wave-uniform edge -> scalar loads
  int lane = t & 31;
  if (ed >= e) return;
  int r = row[ed], c = col[ed];
  float wA = nA[ed], wB = nB[ed];
  float4 v = *(const float4*)(in + r*CDIM + lane*4);
  float* oA = outA + c*CDIM + lane*4;
  float* oB = outB + c*CDIM + lane*4;
  atomicAdd(oA+0, wA*v.x); atomicAdd(oA+1, wA*v.y);
  atomicAdd(oA+2, wA*v.z); atomicAdd(oA+3, wA*v.w);
  atomicAdd(oB+0, wB*v.x); atomicAdd(oB+1, wB*v.y);
  atomicAdd(oB+2, wB*v.z); atomicAdd(oB+3, wB*v.w);
}

// Second-order propagation: out[col[e]] += scale*norm[e]*in[row[e]]
__global__ void k_scatter(const float* __restrict__ in, float* __restrict__ out,
                          const int* __restrict__ row, const int* __restrict__ col,
                          const float* __restrict__ norm, float scale, int e) {
  int t = blockIdx.x*blockDim.x + threadIdx.x;
  int ed = __builtin_amdgcn_readfirstlane(t >> 5);   // wave-uniform edge -> scalar loads
  int lane = t & 31;
  if (ed >= e) return;
  int r = row[ed], c = col[ed];
  float w = scale * norm[ed];
  float4 v = *(const float4*)(in + r*CDIM + lane*4);
  float* o = out + c*CDIM + lane*4;
  atomicAdd(o+0, w*v.x);
  atomicAdd(o+1, w*v.y);
  atomicAdd(o+2, w*v.z);
  atomicAdd(o+3, w*v.w);
}

// ---------------- WMMA GEMM ----------------
// A fragment (contiguous k order used by this code, B matches; not the ISA table order): lane holds row = lane&15, half = lane>>4;
// elems 0..7  <- K = kb + half*16 + 0..7
// elems 8..15 <- K = kb + half*16 + 8..15
__device__ inline v16bf load_a_bf16(const float* __restrict__ A, int rowIdx, int kb, int half) {
  int off = rowIdx*CDIM + kb + half*16;   // 32-bit offset -> saddr+voffset global loads
  float4 f0 = *(const float4*)(A + off + 0);
  float4 f1 = *(const float4*)(A + off + 4);
  float4 f2 = *(const float4*)(A + off + 8);
  float4 f3 = *(const float4*)(A + off + 12);
  v16bf a;
  a[0]=(__bf16)f0.x;  a[1]=(__bf16)f0.y;  a[2]=(__bf16)f0.z;  a[3]=(__bf16)f0.w;
  a[4]=(__bf16)f1.x;  a[5]=(__bf16)f1.y;  a[6]=(__bf16)f1.z;  a[7]=(__bf16)f1.w;
  a[8]=(__bf16)f2.x;  a[9]=(__bf16)f2.y;  a[10]=(__bf16)f2.z; a[11]=(__bf16)f2.w;
  a[12]=(__bf16)f3.x; a[13]=(__bf16)f3.y; a[14]=(__bf16)f3.z; a[15]=(__bf16)f3.w;
  return a;
}

__device__ inline float sigmoidf_(float x) { return 1.f/(1.f + __expf(-x)); }

// Fused Z,R GEMM: D = [A0|A1|A2|A3|A4] @ Wt^T  (K=640), sigmoid epilogue.
__global__ void __launch_bounds__(256)
k_gemm_zr(const float* __restrict__ A0, const float* __restrict__ A1,
          const float* __restrict__ A2, const float* __restrict__ A3,
          const float* __restrict__ A4,
          const __bf16* __restrict__ WtZ, const __bf16* __restrict__ WtR,
          const float* __restrict__ bz, const float* __restrict__ br,
          float* __restrict__ Z, float* __restrict__ R) {
  int wave = blockIdx.x*(blockDim.x >> 5) + (threadIdx.x >> 5);
  if (wave >= NTILES) return;          // uniform per wave; EXEC stays all-ones
  int lane = threadIdx.x & 31;
  int half = lane >> 4, l15 = lane & 15;
  int row0 = wave * 16;
  int rowIdx = row0 + l15;

  v8f accZ[4] = {v8f(0.f), v8f(0.f), v8f(0.f), v8f(0.f)};
  v8f accR[4] = {v8f(0.f), v8f(0.f), v8f(0.f), v8f(0.f)};
  const float* const As[NSRC] = {A0, A1, A2, A3, A4};

#pragma unroll
  for (int s = 0; s < NSRC; ++s) {
    const float* A = As[s];            // resolves to kernel-arg SGPR base after unroll
#pragma unroll
    for (int kc = 0; kc < 4; ++kc) {
      int kb = kc*32;
      v16bf a = load_a_bf16(A, rowIdx, kb, half);
      int skb = s*CDIM + kb + half*16;      // B elem e <-> K = half*16 + e
#pragma unroll
      for (int g = 0; g < 4; ++g) {
        v16bf bzf = *(const v16bf*)(WtZ + (g*16 + l15)*KTOT + skb);
        accZ[g] = __builtin_amdgcn_wmma_f32_16x16x32_bf16(
            false, a, false, bzf, (short)0, accZ[g], false, false);
        v16bf brf = *(const v16bf*)(WtR + (g*16 + l15)*KTOT + skb);
        accR[g] = __builtin_amdgcn_wmma_f32_16x16x32_bf16(
            false, a, false, brf, (short)0, accR[g], false, false);
      }
    }
  }
#pragma unroll
  for (int g = 0; g < 4; ++g) {
    int c = g*16 + l15;
    float bzv = bz[c], brv = br[c];
#pragma unroll
    for (int v = 0; v < 8; ++v) {
      int r = row0 + half*8 + v;          // C/D layout: row = half*8 + vgpr
      int o = r*FDIM + c;
      Z[o] = sigmoidf_(accZ[g][v] + bzv);
      R[o] = sigmoidf_(accR[g][v] + brv);
    }
  }
}

// H-tilde GEMM with fused GRU combine: out = Z*H + (1-Z)*tanh(pre+b)
__global__ void __launch_bounds__(256)
k_gemm_h(const float* __restrict__ A0, const float* __restrict__ A1,
         const float* __restrict__ A2, const float* __restrict__ A3,
         const float* __restrict__ A4,
         const __bf16* __restrict__ WtH, const float* __restrict__ bh,
         const float* __restrict__ Zb, const float* __restrict__ Hin,
         float* __restrict__ Out) {
  int wave = blockIdx.x*(blockDim.x >> 5) + (threadIdx.x >> 5);
  if (wave >= NTILES) return;
  int lane = threadIdx.x & 31;
  int half = lane >> 4, l15 = lane & 15;
  int row0 = wave * 16;
  int rowIdx = row0 + l15;

  v8f acc[4] = {v8f(0.f), v8f(0.f), v8f(0.f), v8f(0.f)};
  const float* const As[NSRC] = {A0, A1, A2, A3, A4};

#pragma unroll
  for (int s = 0; s < NSRC; ++s) {
    const float* A = As[s];
#pragma unroll
    for (int kc = 0; kc < 4; ++kc) {
      int kb = kc*32;
      v16bf a = load_a_bf16(A, rowIdx, kb, half);
      int skb = s*CDIM + kb + half*16;
#pragma unroll
      for (int g = 0; g < 4; ++g) {
        v16bf bf = *(const v16bf*)(WtH + (g*16 + l15)*KTOT + skb);
        acc[g] = __builtin_amdgcn_wmma_f32_16x16x32_bf16(
            false, a, false, bf, (short)0, acc[g], false, false);
      }
    }
  }
#pragma unroll
  for (int g = 0; g < 4; ++g) {
    int c = g*16 + l15;
    float bhv = bh[c];
#pragma unroll
    for (int v = 0; v < 8; ++v) {
      int r = row0 + half*8 + v;
      int o = r*FDIM + c;
      float ht = tanhf(acc[g][v] + bhv);
      float z  = Zb[o];
      Out[o] = z*Hin[o] + (1.f - z)*ht;
    }
  }
}

// ---------------- nanmean infill ----------------

__global__ void k_nan_reduce(const float* __restrict__ x, int n, float* __restrict__ red) {
  __shared__ float ssum[256];
  __shared__ float scnt[256];
  float s = 0.f, c = 0.f;
  for (int i = blockIdx.x*blockDim.x + threadIdx.x; i < n; i += gridDim.x*blockDim.x) {
    float v = x[i];
    if (!(v != v)) { s += v; c += 1.f; }
  }
  ssum[threadIdx.x] = s; scnt[threadIdx.x] = c;
  __syncthreads();
  for (int off = 128; off > 0; off >>= 1) {
    if ((int)threadIdx.x < off) {
      ssum[threadIdx.x] += ssum[threadIdx.x + off];
      scnt[threadIdx.x] += scnt[threadIdx.x + off];
    }
    __syncthreads();
  }
  if (threadIdx.x == 0) { atomicAdd(&red[0], ssum[0]); atomicAdd(&red[1], scnt[0]); }
}

__global__ void k_nan_replace(float* __restrict__ x, int n, const float* __restrict__ red) {
  int i = blockIdx.x*blockDim.x + threadIdx.x;
  if (i < n) {
    float v = x[i];
    if (v != v) {
      float cnt = red[1];
      x[i] = (cnt > 0.f) ? (red[0]/cnt) : v;
    }
  }
}

// ---------------- launch ----------------

extern "C" void kernel_launch(void* const* d_in, const int* in_sizes, int n_in,
                              void* d_out, int out_size, void* d_ws, size_t ws_size,
                              hipStream_t stream) {
  (void)in_sizes; (void)n_in; (void)out_size; (void)ws_size;

  const float* X  = (const float*)d_in[0];
  const int*   ei = (const int*)  d_in[1];
  const float* H  = (const float*)d_in[2];
  const float* Wz = (const float*)d_in[3];
  const float* bz = (const float*)d_in[4];
  const float* Wr = (const float*)d_in[5];
  const float* br = (const float*)d_in[6];
  const float* Wh = (const float*)d_in[7];
  const float* bh = (const float*)d_in[8];
  float* out = (float*)d_out;

  const int* row = ei;
  const int* col = ei + NEDGES;

  // workspace carve (fp32 section then bf16 section; all offsets 32B aligned)
  float* p = (float*)d_ws;
  float* XH   = p; p += (size_t)NNODES*CDIM;
  float* T1o  = p; p += (size_t)NNODES*CDIM;
  float* T1i  = p; p += (size_t)NNODES*CDIM;
  float* T2o  = p; p += (size_t)NNODES*CDIM;
  float* T2i  = p; p += (size_t)NNODES*CDIM;
  float* Zb   = p; p += (size_t)NNODES*FDIM;
  float* Rb   = p; p += (size_t)NNODES*FDIM;
  float* dego = p; p += NNODES;
  float* degi = p; p += NNODES;
  float* nrmo = p; p += NEDGES;
  float* nrmi = p; p += NEDGES;
  float* red  = p; p += 8;
  __bf16* WtZ = (__bf16*)p;
  __bf16* WtR = WtZ + (size_t)FDIM*KTOT;
  __bf16* WtH = WtR + (size_t)FDIM*KTOT;

  const int BT = 256;
  int gN128 = cdiv_host((long long)NNODES*CDIM, BT);
  int gN64  = cdiv_host((long long)NNODES*FDIM, BT);
  int gN    = cdiv_host(NNODES, BT);
  int gE    = cdiv_host(NEDGES, BT);
  int gE32  = cdiv_host((long long)NEDGES*32, BT);
  int gW    = cdiv_host(FDIM*KTOT, BT);
  int gGemm = cdiv_host(NTILES, BT/32);
  int gRed  = 512;

  // degrees + norms
  k_zero2<<<gN, BT, 0, stream>>>(dego, degi, NNODES);
  k_degree<<<gE, BT, 0, stream>>>(row, col, dego, degi, NEDGES);
  k_norm<<<gE, BT, 0, stream>>>(row, dego, degi, nrmo, nrmi, NEDGES);

  // features + weights
  k_build_xh<<<gN64, BT, 0, stream>>>(X, H, XH, NNODES*FDIM);
  k_prep_weights<<<gW, BT, 0, stream>>>(Wz, Wr, Wh, WtZ, WtR, WtH);

  // Chebyshev diffusion on XH (shared by Z and R)
  k_zero2<<<gN128, BT, 0, stream>>>(T1o, T1i, NNODES*CDIM);
  k_scatter2<<<gE32, BT, 0, stream>>>(XH, T1o, T1i, row, col, nrmo, nrmi, NEDGES);
  k_negcopy2<<<gN128, BT, 0, stream>>>(T2o, T2i, XH, NNODES*CDIM);
  k_scatter<<<gE32, BT, 0, stream>>>(T1o, T2o, row, col, nrmo, 2.f, NEDGES);
  k_scatter<<<gE32, BT, 0, stream>>>(T1i, T2i, row, col, nrmi, 2.f, NEDGES);

  // Z, R gates (WMMA, fused sigmoid)
  k_gemm_zr<<<gGemm, BT, 0, stream>>>(XH, T1o, T1i, T2o, T2i, WtZ, WtR, bz, br, Zb, Rb);

  // XHR = concat(X, H*R) reusing XH buffer, then its diffusion
  k_build_xhr<<<gN64, BT, 0, stream>>>(H, Rb, XH, NNODES*FDIM);
  k_zero2<<<gN128, BT, 0, stream>>>(T1o, T1i, NNODES*CDIM);
  k_scatter2<<<gE32, BT, 0, stream>>>(XH, T1o, T1i, row, col, nrmo, nrmi, NEDGES);
  k_negcopy2<<<gN128, BT, 0, stream>>>(T2o, T2i, XH, NNODES*CDIM);
  k_scatter<<<gE32, BT, 0, stream>>>(T1o, T2o, row, col, nrmo, 2.f, NEDGES);
  k_scatter<<<gE32, BT, 0, stream>>>(T1i, T2i, row, col, nrmi, 2.f, NEDGES);

  // H-tilde + GRU combine (WMMA, fused tanh/lerp)
  k_gemm_h<<<gGemm, BT, 0, stream>>>(XH, T1o, T1i, T2o, T2i, WtH, bh, Zb, H, out);

  // nanmean infill (faithful to reference)
  k_zero2<<<1, 32, 0, stream>>>(red, red + 1, 1);
  k_nan_reduce<<<gRed, BT, 0, stream>>>(out, NNODES*FDIM, red);
  k_nan_replace<<<gN64, BT, 0, stream>>>(out, NNODES*FDIM, red);
}
